// GaussianMixture_64965675319481
// MI455X (gfx1250) — matrix operand order repaired and checked by reading, not compiled, not run on hardware
//
#include <hip/hip_runtime.h>
#include <math.h>

// ---------- CDNA5 (gfx1250) wave32 WMMA types ----------
typedef __attribute__((ext_vector_type(16))) __bf16       v16bf;
typedef __attribute__((ext_vector_type(8)))  float        v8f;
typedef __attribute__((ext_vector_type(4)))  float        f32x4;
typedef __attribute__((ext_vector_type(4)))  unsigned int u32x4;

union Frag16 { u32x4 q[2]; v16bf v; };   // two ds_load_b128 -> one bf16 fragment

__device__ __forceinline__ v8f wmma_bf16(v16bf a, v16bf b, v8f c) {
  // (neg_a, A, neg_b, B, c_mod, C, reuse_a, reuse_b)  -> v_wmma_f32_16x16x32_bf16
  return __builtin_amdgcn_wmma_f32_16x16x32_bf16(false, a, false, b, (short)0, c,
                                                 false, false);
}

static constexpr int SLEN   = 65536;
static constexpr int DIM    = 128;
static constexpr int KCOMP  = 8;
static constexpr int NSLICE = 16;

// ---------- workspace layout (bytes, 256-aligned) ----------
static constexpr size_t alup(size_t v) { return (v + 255) & ~(size_t)255; }
static constexpr size_t O_GAMMA = 0;                                           // S*K f32
static constexpr size_t O_MPART = alup(O_GAMMA + (size_t)SLEN*KCOMP*4);        // NSLICE*K*D*D f32
static constexpr size_t O_SXP   = alup(O_MPART + (size_t)NSLICE*KCOMP*DIM*DIM*4);
static constexpr size_t O_NKP   = alup(O_SXP   + (size_t)NSLICE*KCOMP*DIM*4);
static constexpr size_t O_NK    = alup(O_NKP   + (size_t)NSLICE*KCOMP*4);
static constexpr size_t O_MU    = alup(O_NK    + (size_t)KCOMP*4);
static constexpr size_t O_SIG   = alup(O_MU    + (size_t)KCOMP*DIM*4);
static constexpr size_t O_Y     = alup(O_SIG   + (size_t)KCOMP*DIM*DIM*4);     // L^-1 scratch
static constexpr size_t O_BBF   = alup(O_Y     + (size_t)KCOMP*DIM*DIM*4);     // bf16 Sigma^-1
static constexpr size_t O_BMU   = alup(O_BBF   + (size_t)KCOMP*DIM*DIM*2);
static constexpr size_t O_CK    = alup(O_BMU   + (size_t)KCOMP*DIM*4);
static constexpr size_t O_LC    = alup(O_CK    + (size_t)KCOMP*4);
static constexpr size_t O_EP    = alup(O_LC    + (size_t)KCOMP*4);             // SLEN/64 f32

// ================================================================
// K1: gamma = softmax(x @ W + b)    [memory-bound, reads x once]
// ================================================================
__global__ __launch_bounds__(256) void k_gamma(const float* __restrict__ x,
                                               const float* __restrict__ W,
                                               const float* __restrict__ b,
                                               float* __restrict__ gamma) {
  __shared__ float Wl[DIM * KCOMP];
  __shared__ float bl[KCOMP];
  const int t = threadIdx.x;
  for (int i = t; i < DIM * KCOMP; i += 256) Wl[i] = W[i];
  if (t < KCOMP) bl[t] = b[t];
  __syncthreads();
  const size_t s = (size_t)blockIdx.x * 256 + t;
  const f32x4* xr = (const f32x4*)(x + s * DIM);
  float lg[KCOMP];
#pragma unroll
  for (int k = 0; k < KCOMP; k++) lg[k] = bl[k];
#pragma unroll 4
  for (int q = 0; q < DIM / 4; q++) {
    f32x4 v = xr[q];
    const int d = q * 4;
#pragma unroll
    for (int k = 0; k < KCOMP; k++)
      lg[k] += v.x * Wl[(d + 0) * KCOMP + k] + v.y * Wl[(d + 1) * KCOMP + k] +
               v.z * Wl[(d + 2) * KCOMP + k] + v.w * Wl[(d + 3) * KCOMP + k];
  }
  float mx = lg[0];
#pragma unroll
  for (int k = 1; k < KCOMP; k++) mx = fmaxf(mx, lg[k]);
  float sum = 0.0f;
#pragma unroll
  for (int k = 0; k < KCOMP; k++) { lg[k] = expf(lg[k] - mx); sum += lg[k]; }
  const float inv = 1.0f / sum;
  float* gp = gamma + s * KCOMP;
#pragma unroll
  for (int k = 0; k < KCOMP; k++) gp[k] = lg[k] * inv;
}

// ================================================================
// K2: M_k = sum_s gamma_sk x_s x_s^T  via v_wmma_f32_16x16x32_bf16
//     one workgroup = (component k, S-slice); 8 waves; wave m owns
//     the 16-row band d=16m..16m+15 of the 128x128 output.
//     Double-buffered LDS staging (1 barrier / 32-sample chunk,
//     global loads overlap the WMMA stream); all 8 B fragments are
//     preloaded so the 8 WMMAs issue back-to-back after one
//     s_wait_dscnt.  Also produces deterministic Sx / Nk partials.
// ================================================================
__global__ __launch_bounds__(256) void k_cov(const float* __restrict__ x,
                                             const float* __restrict__ gamma,
                                             float* __restrict__ Mpart,
                                             float* __restrict__ Sxpart,
                                             float* __restrict__ Nkpart) {
  const int k = blockIdx.x & (KCOMP - 1);
  const int slice = blockIdx.x >> 3;
  const int t = threadIdx.x;
  const int lane = t & 31, wave = t >> 5;
  const int hi = lane >> 4, lq = lane & 15;
  const int ss = t >> 3, dg = t & 7;           // staging: sample-slot, d-group

  __shared__ alignas(16) __bf16 xt[2][DIM * 32];   // [d][ss]  B-operand (plain x)
  __shared__ alignas(16) __bf16 gxt[2][DIM * 32];  // [d][ss]  A-operand (gamma*x)
  __shared__ float redbuf[32 * DIM];
  __shared__ float nkl[32];

  const v8f vz = {0, 0, 0, 0, 0, 0, 0, 0};
  v8f acc[8];
#pragma unroll
  for (int n = 0; n < 8; n++) acc[n] = vz;
  float sxacc[16];
#pragma unroll
  for (int j = 0; j < 16; j++) sxacc[j] = 0.0f;
  float nkacc = 0.0f;

  const int spw = SLEN / NSLICE;  // samples per workgroup
  const int sbase = slice * spw;
  const int nch = spw / 32;

  // stage chunk c into LDS buffer buf (coalesced f32 reads, bf16 transpose)
  auto stage = [&](int c, __bf16* xtb, __bf16* gxtb) {
    const int s0 = sbase + c * 32;
    const f32x4* xr = (const f32x4*)(x + (size_t)(s0 + ss) * DIM + dg * 16);
    const float g = gamma[(size_t)(s0 + ss) * KCOMP + k];
    if (dg == 0) nkacc += g;
#pragma unroll
    for (int q = 0; q < 4; q++) {
      f32x4 v = xr[q];
      const int d = dg * 16 + q * 4;
      xtb[(d + 0) * 32 + ss] = (__bf16)v.x;  xtb[(d + 1) * 32 + ss] = (__bf16)v.y;
      xtb[(d + 2) * 32 + ss] = (__bf16)v.z;  xtb[(d + 3) * 32 + ss] = (__bf16)v.w;
      const float g0 = v.x * g, g1 = v.y * g, g2 = v.z * g, g3 = v.w * g;
      gxtb[(d + 0) * 32 + ss] = (__bf16)g0;  gxtb[(d + 1) * 32 + ss] = (__bf16)g1;
      gxtb[(d + 2) * 32 + ss] = (__bf16)g2;  gxtb[(d + 3) * 32 + ss] = (__bf16)g3;
      sxacc[q * 4 + 0] += g0; sxacc[q * 4 + 1] += g1;
      sxacc[q * 4 + 2] += g2; sxacc[q * 4 + 3] += g3;
    }
  };

  stage(0, xt[0], gxt[0]);
  for (int c = 0; c < nch; c++) {
    __syncthreads();                 // buffer cur fully staged; prev compute done
    const int cur = c & 1;
    if (c + 1 < nch) stage(c + 1, xt[cur ^ 1], gxt[cur ^ 1]);
    // ---- A fragment: rows d = 16*wave..+15, K = 32 samples ----
    Frag16 a;
    const __bf16* ap = gxt[cur] + (wave * 16 + lq) * 32 + hi * 8;
    a.q[0] = *(const u32x4*)ap;
    a.q[1] = *(const u32x4*)(ap + 16);
    // ---- preload all 8 B fragments, then WMMAs back-to-back ----
    Frag16 bm[8];
#pragma unroll
    for (int n = 0; n < 8; n++) {
      const __bf16* bp = xt[cur] + (n * 16 + lq) * 32 + hi * 8;
      bm[n].q[0] = *(const u32x4*)bp;
      bm[n].q[1] = *(const u32x4*)(bp + 16);
    }
#pragma unroll
    for (int n = 0; n < 8; n++) acc[n] = wmma_bf16(a.v, bm[n].v, acc[n]);
  }
  // ---- write 128x128 partial (C layout: VGPR j -> row j + 8*hi) ----
  float* Mout = Mpart + ((size_t)slice * KCOMP + k) * DIM * DIM;
#pragma unroll
  for (int n = 0; n < 8; n++) {
#pragma unroll
    for (int j = 0; j < 8; j++) {
      const int d = wave * 16 + j + hi * 8;
      const int e = n * 16 + lq;
      Mout[d * DIM + e] = acc[n][j];
    }
  }
  // ---- deterministic Sx / Nk partials ----
#pragma unroll
  for (int j = 0; j < 16; j++) redbuf[ss * DIM + dg * 16 + j] = sxacc[j];
  if (dg == 0) nkl[ss] = nkacc;
  __syncthreads();
  if (t < DIM) {
    float sxs = 0.0f;
    for (int i = 0; i < 32; i++) sxs += redbuf[i * DIM + t];
    Sxpart[((size_t)slice * KCOMP + k) * DIM + t] = sxs;
  }
  if (t == 0) {
    float nk = 0.0f;
    for (int i = 0; i < 32; i++) nk += nkl[i];
    Nkpart[slice * KCOMP + k] = nk;
  }
}

// ================================================================
// K3a: reduce slices -> Nk, mu, Sigma = M/Nk - mu mu^T
// ================================================================
__global__ __launch_bounds__(256) void k_sigma(const float* __restrict__ Mpart,
                                               const float* __restrict__ Sxpart,
                                               const float* __restrict__ Nkpart,
                                               float* __restrict__ NkOut,
                                               float* __restrict__ muOut,
                                               float* __restrict__ sigma) {
  const int k = blockIdx.x, t = threadIdx.x;
  __shared__ float mus[DIM];
  __shared__ float nk_s;
  if (t == 0) {
    float nk = 0.0f;
    for (int sl = 0; sl < NSLICE; sl++) nk += Nkpart[sl * KCOMP + k];
    nk_s = nk; NkOut[k] = nk;
  }
  __syncthreads();
  if (t < DIM) {
    float sx = 0.0f;
    for (int sl = 0; sl < NSLICE; sl++) sx += Sxpart[((size_t)sl * KCOMP + k) * DIM + t];
    const float m = sx / nk_s;
    mus[t] = m; muOut[k * DIM + t] = m;
  }
  __syncthreads();
  const float inv = 1.0f / nk_s;
  for (int idx = t; idx < DIM * DIM; idx += 256) {
    float m = 0.0f;
    for (int sl = 0; sl < NSLICE; sl++)
      m += Mpart[((size_t)(sl * KCOMP + k)) * DIM * DIM + idx];
    const int d = idx >> 7, e = idx & 127;
    sigma[(size_t)k * DIM * DIM + idx] = m * inv - mus[d] * mus[e];
  }
}

// ================================================================
// K3b: per-component Cholesky -> L^-1 -> B = Sigma^-1 (bf16),
//      logdet, Bmu, mu^T B mu, log-coefficient.  O(D^3) ~ noise.
// ================================================================
__global__ __launch_bounds__(128) void k_chol(const float* __restrict__ sigma,
                                              const float* __restrict__ Nk,
                                              const float* __restrict__ mu,
                                              float* __restrict__ Yg,
                                              __bf16* __restrict__ Bbf,
                                              float* __restrict__ Bmu,
                                              float* __restrict__ ckA,
                                              float* __restrict__ lcA) {
  const int k = blockIdx.x, t = threadIdx.x;
  __shared__ float A[DIM * DIM];
  __shared__ float mus[DIM];
  __shared__ float redA[DIM], redB[DIM];
  const float* Sg = sigma + (size_t)k * DIM * DIM;
  for (int i = t; i < DIM * DIM; i += 128) A[i] = Sg[i];
  mus[t] = mu[k * DIM + t];
  __syncthreads();
  // right-looking Cholesky, thread t owns row t of the lower triangle
  for (int j = 0; j < DIM; j++) {
    if (t == 0) A[j * DIM + j] = sqrtf(fmaxf(A[j * DIM + j], 1e-30f));
    __syncthreads();
    const float pinv = 1.0f / A[j * DIM + j];
    if (t > j) A[t * DIM + j] *= pinv;
    __syncthreads();
    if (t > j) {
      const float lij = A[t * DIM + j];
      for (int l = j + 1; l <= t; l++) A[t * DIM + l] -= lij * A[l * DIM + j];
    }
    __syncthreads();
  }
  redA[t] = logf(A[t * DIM + t]);
  // forward substitution: column t of Y = L^-1 (column private to thread)
  float* Y = Yg + (size_t)k * DIM * DIM;
  for (int r = t; r < DIM; r++) {
    float v = (r == t) ? 1.0f : 0.0f;
    for (int p = t; p < r; p++) v -= A[r * DIM + p] * Y[p * DIM + t];
    Y[r * DIM + t] = v / A[r * DIM + r];
  }
  __syncthreads();
  // B = Y^T Y (symmetric); thread t computes row d = t
  float bmuacc = 0.0f;
  for (int e = 0; e < DIM; e++) {
    const int lo = (t > e) ? t : e;
    float s = 0.0f;
    for (int r = lo; r < DIM; r++) s += Y[r * DIM + t] * Y[r * DIM + e];
    Bbf[(size_t)k * DIM * DIM + t * DIM + e] = (__bf16)s;
    bmuacc += s * mus[e];
  }
  Bmu[k * DIM + t] = bmuacc;
  redB[t] = bmuacc * mus[t];
  __syncthreads();
  if (t == 0) {
    float ld = 0.0f, ck = 0.0f;
    for (int i = 0; i < DIM; i++) { ld += redA[i]; ck += redB[i]; }
    ld *= 2.0f;
    ckA[k] = ck;
    const float phi = Nk[k] / (float)SLEN;
    lcA[k] = logf(phi) - 0.5f * ((float)DIM * logf(6.28318530717958647692f) + ld);
  }
}

// ================================================================
// K4: maha_sk = x^T B x - 2 x^T(B mu) + mu^T B mu  via WMMA
//     block = 64 samples, 4 waves; wave m owns sample band 16m..16m+15.
//     B fragments preloaded per 32-wide K step -> 8 WMMAs back-to-back.
//     dens_s = sum_k exp(logcoef_k - maha/2); partial sum(log) per block.
// ================================================================
__global__ __launch_bounds__(128) void k_dens(const float* __restrict__ x,
                                              const u32x4* __restrict__ Bbfq,
                                              const float* __restrict__ Bmu,
                                              const float* __restrict__ ckA,
                                              const float* __restrict__ lcA,
                                              float* __restrict__ epart) {
  __shared__ alignas(16) __bf16 xbf[64 * DIM];   // [s][d], 256B rows
  __shared__ alignas(16) __bf16 Bl[DIM * DIM];   // [e][d] == B (symmetric)
  __shared__ float bmus[DIM];
  __shared__ float qbuf[64];
  const int t = threadIdx.x;
  const int lane = t & 31, wave = t >> 5;
  const int hi = lane >> 4, lq = lane & 15;
  const int s0 = blockIdx.x * 64;
  {
    const int ss = t >> 1, dh = t & 1;
    const f32x4* xr = (const f32x4*)(x + (size_t)(s0 + ss) * DIM + dh * 64);
#pragma unroll
    for (int q = 0; q < 16; q++) {
      f32x4 v = xr[q];
      const int d = dh * 64 + q * 4;
      xbf[ss * DIM + d + 0] = (__bf16)v.x;  xbf[ss * DIM + d + 1] = (__bf16)v.y;
      xbf[ss * DIM + d + 2] = (__bf16)v.z;  xbf[ss * DIM + d + 3] = (__bf16)v.w;
    }
  }
  float dens = 0.0f;
  const v8f vz = {0, 0, 0, 0, 0, 0, 0, 0};
  for (int k = 0; k < KCOMP; k++) {
    __syncthreads();  // protect Bl/bmus/qbuf from previous iteration
    {
      u32x4* dst = (u32x4*)Bl;
      const u32x4* src = Bbfq + (size_t)k * (DIM * DIM / 8);
      for (int i = t; i < DIM * DIM / 8; i += 128) dst[i] = src[i];
      bmus[t] = Bmu[k * DIM + t];
    }
    __syncthreads();
    v8f acc[8];
#pragma unroll
    for (int n = 0; n < 8; n++) acc[n] = vz;
#pragma unroll
    for (int kc = 0; kc < 4; kc++) {          // K-dim (d) in chunks of 32
      Frag16 a;
      const __bf16* ap = xbf + (wave * 16 + lq) * DIM + kc * 32 + hi * 8;
      a.q[0] = *(const u32x4*)ap;
      a.q[1] = *(const u32x4*)(ap + 16);
      Frag16 bm[8];
#pragma unroll
      for (int n = 0; n < 8; n++) {
        const __bf16* bp = Bl + (n * 16 + lq) * DIM + kc * 32 + hi * 8;
        bm[n].q[0] = *(const u32x4*)bp;
        bm[n].q[1] = *(const u32x4*)(bp + 16);
      }
#pragma unroll
      for (int n = 0; n < 8; n++) acc[n] = wmma_bf16(a.v, bm[n].v, acc[n]);
    }
    // q_s = row_s(Z) . x_s   (Z lives in the accumulators)
    float q8[8];
#pragma unroll
    for (int j = 0; j < 8; j++) q8[j] = 0.0f;
#pragma unroll
    for (int n = 0; n < 8; n++) {
#pragma unroll
      for (int j = 0; j < 8; j++) {
        const int srow = wave * 16 + j + hi * 8;
        q8[j] += acc[n][j] * (float)xbf[srow * DIM + n * 16 + lq];
      }
    }
#pragma unroll
    for (int j = 0; j < 8; j++) {            // reduce over the 16 lanes of each half
      float v = q8[j];
      v += __shfl_xor(v, 1, 32);
      v += __shfl_xor(v, 2, 32);
      v += __shfl_xor(v, 4, 32);
      v += __shfl_xor(v, 8, 32);
      if (lq == 0) qbuf[wave * 16 + j + hi * 8] = v;
    }
    __syncthreads();
    if (t < 64) {
      float xd = 0.0f;
      for (int d = 0; d < DIM; d++) xd += (float)xbf[t * DIM + d] * bmus[d];
      const float maha = qbuf[t] - 2.0f * xd + ckA[k];
      dens += expf(lcA[k] - 0.5f * maha);
    }
  }
  __syncthreads();
  if (t < 64) qbuf[t] = logf(dens);
  __syncthreads();
  if (t == 0) {
    float s = 0.0f;
    for (int i = 0; i < 64; i++) s += qbuf[i];
    epart[blockIdx.x] = s;
  }
}

// ================================================================
// K5: energy = -sum(log dens)  (deterministic tree reduction)
// ================================================================
__global__ __launch_bounds__(256) void k_final(const float* __restrict__ epart,
                                               float* __restrict__ out) {
  __shared__ float red[256];
  const int t = threadIdx.x;
  float s = 0.0f;
  for (int i = t; i < SLEN / 64; i += 256) s += epart[i];
  red[t] = s;
  __syncthreads();
  for (int w = 128; w > 0; w >>= 1) {
    if (t < w) red[t] += red[t + w];
    __syncthreads();
  }
  if (t == 0) out[0] = -red[0];
}

// ================================================================
extern "C" void kernel_launch(void* const* d_in, const int* in_sizes, int n_in,
                              void* d_out, int out_size, void* d_ws, size_t ws_size,
                              hipStream_t stream) {
  (void)in_sizes; (void)n_in; (void)out_size; (void)ws_size;
  const float* x = (const float*)d_in[0];
  const float* W = (const float*)d_in[1];
  const float* b = (const float*)d_in[2];
  char* ws = (char*)d_ws;
  float*  gamma  = (float*)(ws + O_GAMMA);
  float*  Mpart  = (float*)(ws + O_MPART);
  float*  Sxpart = (float*)(ws + O_SXP);
  float*  Nkpart = (float*)(ws + O_NKP);
  float*  Nk     = (float*)(ws + O_NK);
  float*  mu     = (float*)(ws + O_MU);
  float*  sigma  = (float*)(ws + O_SIG);
  float*  Yg     = (float*)(ws + O_Y);
  __bf16* Bbf    = (__bf16*)(ws + O_BBF);
  float*  Bmu    = (float*)(ws + O_BMU);
  float*  ckA    = (float*)(ws + O_CK);
  float*  lcA    = (float*)(ws + O_LC);
  float*  epart  = (float*)(ws + O_EP);

  k_gamma<<<SLEN / 256, 256, 0, stream>>>(x, W, b, gamma);
  k_cov<<<NSLICE * KCOMP, 256, 0, stream>>>(x, gamma, Mpart, Sxpart, Nkpart);
  k_sigma<<<KCOMP, 256, 0, stream>>>(Mpart, Sxpart, Nkpart, Nk, mu, sigma);
  k_chol<<<KCOMP, 128, 0, stream>>>(sigma, Nk, mu, Yg, Bbf, Bmu, ckA, lcA);
  k_dens<<<SLEN / 64, 128, 0, stream>>>(x, (const u32x4*)Bbf, Bmu, ckA, lcA, epart);
  k_final<<<1, 256, 0, stream>>>(epart, (float*)d_out);
}
